// InternLM2MambaDecoderLayer_27882927685977
// MI455X (gfx1250) — compile-verified, matrix operand order checked by reading, not run on hardware
//
#include <hip/hip_runtime.h>

// ---------------------------------------------------------------- types
typedef __attribute__((ext_vector_type(16))) __bf16 v16bf;
typedef __attribute__((ext_vector_type(8)))  float  v8f;

union Frag {
    v16bf v;
    uint4 q[2];
};

#define V8F_ZERO ((v8f){0.f,0.f,0.f,0.f,0.f,0.f,0.f,0.f})

// ---------------------------------------------------------------- constants
#define SEQ      2048
#define BATCH    2
#define NTOK     4096           // BATCH*SEQ
#define DMODEL   2048
#define DINNER   2048
#define DXB      1024
#define DSTATE   128
#define NHEADS   16
#define NKV      8
#define CHUNK    128
#define NCHUNK   16             // SEQ/CHUNK
#define INTER    8192
#define PROJ     6160           // 2*DINNER + 2*DXB + NHEADS
#define CONVDIM  4096           // 2*DXB + DINNER

// GEMM tiling
#define BM 128
#define BN 128
#define BK 32
#define LDT 40                  // GEMM LDS row stride (halves): 32 + 8 pad, 80B (16B aligned)
#define SLDT 136                // SSD LDS row stride (halves): 128 + 8 pad, 272B (16B aligned)

// ---------------------------------------------------------------- helpers
__device__ inline unsigned short f2bf(float f) {
    union { float f; unsigned int u; } v; v.f = f;
    unsigned int u = v.u;
    unsigned int r = u + 0x7FFFu + ((u >> 16) & 1u);
    return (unsigned short)(r >> 16);
}
__device__ inline float bf2f(unsigned short h) {
    union { unsigned int u; float f; } v; v.u = ((unsigned int)h) << 16;
    return v.f;
}
__device__ inline float siluf(float x) { return x / (1.0f + __expf(-x)); }

// CDNA5 async global->LDS copy (ASYNCcnt-tracked, no VGPR round trip).
// LDS byte offset = low 32 bits of the generic pointer (flat aperture rule).
__device__ inline void async_g2l_b128(unsigned lds_off, const void* gaddr) {
    asm volatile("global_load_async_to_lds_b128 %0, %1, off"
                 :: "v"(lds_off), "v"(gaddr) : "memory");
}
__device__ inline void wait_async0() {
    asm volatile("s_wait_asynccnt 0x0" ::: "memory");
}

// load a 16x32 bf16 WMMA operand fragment from a row-major LDS tile
__device__ inline Frag ldsfrag(const unsigned short* base, int row, int stride,
                               int ks, int kh) {
    Frag f;
    f.q[0] = *reinterpret_cast<const uint4*>(&base[row * stride + ks * 32 + kh * 8]);
    f.q[1] = *reinterpret_cast<const uint4*>(&base[row * stride + ks * 32 + 16 + kh * 8]);
    return f;
}

// ---------------------------------------------------------------- weight fp32 -> bf16 transpose (out[N][K])
__global__ __launch_bounds__(256) void k_conv_t_bf16(
    const float* __restrict__ in, unsigned short* __restrict__ out, int K, int N)
{
    __shared__ float tile[32][33];
    int nx = blockIdx.x * 32 + threadIdx.x;  // N index
    int ky = blockIdx.y * 32 + threadIdx.y;  // K base
#pragma unroll
    for (int j = 0; j < 32; j += 8) {
        int k = ky + j;
        tile[threadIdx.y + j][threadIdx.x] =
            (nx < N && k < K) ? in[(long)k * N + nx] : 0.0f;
    }
    __syncthreads();
    int on = blockIdx.x * 32 + threadIdx.y;  // out row (N)
    int ok = blockIdx.y * 32 + threadIdx.x;  // out col (K)
#pragma unroll
    for (int j = 0; j < 32; j += 8) {
        int n = on + j;
        if (n < N && ok < K)
            out[(long)n * K + ok] = f2bf(tile[threadIdx.x][threadIdx.y + j]);
    }
}

// ---------------------------------------------------------------- rmsnorm -> bf16
__global__ __launch_bounds__(256) void k_rmsnorm_bf16(
    const float* __restrict__ x, const float* __restrict__ w,
    unsigned short* __restrict__ out, int cols)
{
    int row = blockIdx.x;
    const float* xr = x + (long)row * cols;
    float s = 0.0f;
    for (int i = threadIdx.x; i < cols; i += 256) { float v = xr[i]; s += v * v; }
    __shared__ float sb[256];
    sb[threadIdx.x] = s; __syncthreads();
    for (int o = 128; o > 0; o >>= 1) {
        if (threadIdx.x < o) sb[threadIdx.x] += sb[threadIdx.x + o];
        __syncthreads();
    }
    float scale = rsqrtf(sb[0] / (float)cols + 1e-5f);
    unsigned short* orow = out + (long)row * cols;
    for (int i = threadIdx.x; i < cols; i += 256)
        orow[i] = f2bf(xr[i] * scale * w[i]);
}

// ---------------------------------------------------------------- gated rmsnorm: t = y*silu(z); rmsnorm(t)*w -> bf16
__global__ __launch_bounds__(256) void k_gatenorm(
    const float* __restrict__ y, const float* __restrict__ zx,
    const float* __restrict__ w, unsigned short* __restrict__ out)
{
    int row = blockIdx.x;
    __shared__ float tb[DINNER];
    __shared__ float sb[256];
    const float* yr = y + (long)row * DINNER;
    const float* zr = zx + (long)row * PROJ;     // z = first DINNER cols of zxbcdt
    float s = 0.0f;
    for (int i = threadIdx.x; i < DINNER; i += 256) {
        float t = yr[i] * siluf(zr[i]);
        tb[i] = t; s += t * t;
    }
    sb[threadIdx.x] = s; __syncthreads();
    for (int o = 128; o > 0; o >>= 1) {
        if (threadIdx.x < o) sb[threadIdx.x] += sb[threadIdx.x + o];
        __syncthreads();
    }
    float scale = rsqrtf(sb[0] / (float)DINNER + 1e-5f);
    unsigned short* orow = out + (long)row * DINNER;
    for (int i = threadIdx.x; i < DINNER; i += 256)
        orow[i] = f2bf(tb[i] * scale * w[i]);
}

// ---------------------------------------------------------------- bf16 WMMA GEMM: C[M,N] = A[M,K] * Bt[N,K]^T
// Staging via CDNA5 global_load_async_to_lds_b128; compute via v_wmma_f32_16x16x32_bf16.
__global__ __launch_bounds__(256) void k_gemm_bf16(
    const unsigned short* __restrict__ A, const unsigned short* __restrict__ Bt,
    float* __restrict__ C, int M, int N, int K)
{
    __shared__ unsigned short As[2][BM * LDT];
    __shared__ unsigned short Bs[2][BM * LDT];

    const int tid  = threadIdx.x;
    const int lane = tid & 31;
    const int wave = tid >> 5;
    const int wm = wave & 3;        // 0..3 : rows   (32 each)
    const int wn = wave >> 2;       // 0..1 : cols   (64 each)
    const int m16 = lane & 15;
    const int kh  = lane >> 4;      // 0/1

    const int row0 = blockIdx.y * BM;
    const int col0 = blockIdx.x * BN;

    const int lr = tid >> 2;        // 0..63
    const int lc = (tid & 3) * 8;   // 0,8,16,24

    // B rows clamped: row n feeds only output column n; epilogue masks gc < N,
    // so clamped (duplicate) rows never reach memory.
    int bn0 = col0 + lr;      if (bn0 > N - 1) bn0 = N - 1;
    int bn1 = col0 + lr + 64; if (bn1 > N - 1) bn1 = N - 1;

    const unsigned aoff0 = (unsigned)(unsigned long long)&As[0][lr * LDT + lc];
    const unsigned aoff1 = (unsigned)(unsigned long long)&As[0][(lr + 64) * LDT + lc];
    const unsigned boff0 = (unsigned)(unsigned long long)&Bs[0][lr * LDT + lc];
    const unsigned boff1 = (unsigned)(unsigned long long)&Bs[0][(lr + 64) * LDT + lc];
    const unsigned bufstep = (unsigned)(BM * LDT * sizeof(unsigned short));

    auto stage = [&](int kt, int buf) {
        const unsigned short* ap = A + (long)(row0 + lr) * K + kt * BK + lc;
        unsigned bo = buf ? bufstep : 0u;
        async_g2l_b128(aoff0 + bo, ap);
        async_g2l_b128(aoff1 + bo, ap + (long)64 * K);
        async_g2l_b128(boff0 + bo, Bt + (long)bn0 * K + kt * BK + lc);
        async_g2l_b128(boff1 + bo, Bt + (long)bn1 * K + kt * BK + lc);
    };

    v8f acc[2][4];
#pragma unroll
    for (int i = 0; i < 2; ++i)
#pragma unroll
        for (int j = 0; j < 4; ++j)
            acc[i][j] = V8F_ZERO;

    const int nk = K / BK;
    stage(0, 0);
    wait_async0();
    __syncthreads();

    for (int kt = 0; kt < nk; ++kt) {
        const int cur = kt & 1;
        if (kt + 1 < nk) stage(kt + 1, cur ^ 1);
        if (kt + 2 < nk) {   // L2 prefetch two tiles ahead (global_prefetch_b8)
            __builtin_prefetch(A + (long)(row0 + lr) * K + (kt + 2) * BK + lc, 0, 1);
            __builtin_prefetch(Bt + (long)bn0 * K + (kt + 2) * BK + lc, 0, 1);
        }

        Frag a[2], b[4];
#pragma unroll
        for (int mt = 0; mt < 2; ++mt)
            a[mt] = ldsfrag(As[cur], wm * 32 + mt * 16 + m16, LDT, 0, kh);
#pragma unroll
        for (int nt = 0; nt < 4; ++nt)
            b[nt] = ldsfrag(Bs[cur], wn * 64 + nt * 16 + m16, LDT, 0, kh);
#pragma unroll
        for (int mt = 0; mt < 2; ++mt)
#pragma unroll
            for (int nt = 0; nt < 4; ++nt)
                acc[mt][nt] = __builtin_amdgcn_wmma_f32_16x16x32_bf16(
                    false, a[mt].v, false, b[nt].v, (short)0, acc[mt][nt], false, false);

        wait_async0();       // next tile fully in LDS
        __syncthreads();
    }

    // epilogue: lane (m16,kh); vgpr v -> row = v + 8*kh, col = m16
#pragma unroll
    for (int mt = 0; mt < 2; ++mt)
#pragma unroll
        for (int nt = 0; nt < 4; ++nt) {
            int gc = col0 + wn * 64 + nt * 16 + m16;
            if (gc < N) {
                long rb = (long)(row0 + wm * 32 + mt * 16 + kh * 8);
#pragma unroll
                for (int v = 0; v < 8; ++v)
                    C[(rb + v) * N + gc] = acc[mt][nt][v];
            }
        }
}

// ---------------------------------------------------------------- causal conv(4) + SiLU over xBC cols of zxbcdt
__global__ __launch_bounds__(256) void k_conv_silu(
    const float* __restrict__ zx, const float* __restrict__ cw,
    const float* __restrict__ cb, float* __restrict__ out)
{
    long idx = (long)blockIdx.x * 256 + threadIdx.x;
    if (idx >= (long)NTOK * CONVDIM) return;
    int c = (int)(idx & (CONVDIM - 1));
    long tok = idx >> 12;
    int t = (int)(tok & (SEQ - 1));
    float acc = cb[c];
#pragma unroll
    for (int k = 0; k < 4; ++k) {
        int tt = t - 3 + k;
        if (tt >= 0)
            acc += zx[(tok - 3 + k) * (long)PROJ + DINNER + c] * cw[k * CONVDIM + c];
    }
    out[idx] = siluf(acc);
}

// ---------------------------------------------------------------- dt = softplus(raw + bias); dA = -exp(A_log)*dt
__global__ __launch_bounds__(256) void k_dtprep(
    const float* __restrict__ zx, const float* __restrict__ dt_bias,
    const float* __restrict__ A_log, float* __restrict__ dt, float* __restrict__ dA)
{
    int idx = blockIdx.x * 256 + threadIdx.x;
    if (idx >= NTOK * NHEADS) return;
    int h = idx & (NHEADS - 1);
    long tok = idx >> 4;
    float x = zx[tok * (long)PROJ + (2 * DINNER + 2 * DXB) + h] + dt_bias[h];
    float sp = (x > 20.0f) ? x : log1pf(__expf(x));
    dt[idx] = sp;
    dA[idx] = -__expf(A_log[h]) * sp;
}

// ---------------------------------------------------------------- SSD: per-chunk states via WMMA
// states[p,n] = sum_s x[s,p]*wdec[s]*B[s,n]  ==  tilemul(Xt[p][s], Bnw[n][s])
// grid (NCHUNK, NHEADS, BATCH); dyn LDS: 2*128*SLDT bf16 = 68 KB
__global__ __launch_bounds__(256) void k_ssd_states(
    const float* __restrict__ xBC, const float* __restrict__ dt,
    const float* __restrict__ dA, float* __restrict__ states,
    float* __restrict__ dAcs)
{
    int c = blockIdx.x, h = blockIdx.y, b = blockIdx.z;
    int kv = h >> 1;
    extern __shared__ unsigned short smu[];
    unsigned short* Xt  = smu;               // [p][s]
    unsigned short* Bnw = smu + 128 * SLDT;  // [n][s], wdec*dt folded
    __shared__ float dAl[CHUNK], cs[CHUNK], wdec[CHUNK];

    const int tid = threadIdx.x;
    long tok0 = (long)b * SEQ + (long)c * CHUNK;
    long bhc  = ((long)(b * NHEADS + h) * NCHUNK + c);

    if (tid < CHUNK) dAl[tid] = dA[(tok0 + tid) * NHEADS + h];
    __syncthreads();
    if (tid == 0) {
        float a = 0.0f;
        for (int s = 0; s < CHUNK; ++s) { a += dAl[s]; cs[s] = a; }
    }
    __syncthreads();
    float total = cs[CHUNK - 1];
    if (tid < CHUNK) {
        wdec[tid] = __expf(total - cs[tid]) * dt[(tok0 + tid) * NHEADS + h];
        dAcs[bhc * CHUNK + tid] = cs[tid];
    }
    __syncthreads();
    for (int i = tid; i < 16384; i += 256) {
        int s = i >> 7, n = i & 127;   // n doubles as p for Xt
        Xt[n * SLDT + s]  = f2bf(xBC[(tok0 + s) * (long)CONVDIM + kv * DSTATE + n]);
        Bnw[n * SLDT + s] = f2bf(xBC[(tok0 + s) * (long)CONVDIM + DXB + kv * DSTATE + n] * wdec[s]);
    }
    __syncthreads();

    const int lane = tid & 31, wave = tid >> 5;
    const int m16 = lane & 15, kh = lane >> 4;
    const int prow = wave * 16 + m16;

    Frag a4[4];
#pragma unroll
    for (int ks = 0; ks < 4; ++ks) a4[ks] = ldsfrag(Xt, prow, SLDT, ks, kh);

    long sbase = bhc * 16384;
    for (int j = 0; j < 8; ++j) {
        v8f acc = V8F_ZERO;
#pragma unroll
        for (int ks = 0; ks < 4; ++ks) {
            Frag bf = ldsfrag(Bnw, j * 16 + m16, SLDT, ks, kh);
            acc = __builtin_amdgcn_wmma_f32_16x16x32_bf16(
                false, a4[ks].v, false, bf.v, (short)0, acc, false, false);
        }
#pragma unroll
        for (int v = 0; v < 8; ++v)
            states[sbase + (long)(wave * 16 + v + 8 * kh) * 128 + j * 16 + m16] = acc[v];
    }
}

// ---------------------------------------------------------------- SSD: inter-chunk scan (16 steps)
__global__ __launch_bounds__(256) void k_ssd_scan(
    const float* __restrict__ states, const float* __restrict__ dAcs,
    float* __restrict__ prev)
{
    int bh = blockIdx.x;  // 0..BATCH*NHEADS-1
    __shared__ float dec[NCHUNK];
    if (threadIdx.x < NCHUNK)
        dec[threadIdx.x] = __expf(dAcs[((long)bh * NCHUNK + threadIdx.x) * CHUNK + (CHUNK - 1)]);
    __syncthreads();
    for (int i = threadIdx.x; i < 16384; i += 256) {
        float P = 0.0f;
#pragma unroll
        for (int c = 0; c < NCHUNK; ++c) {
            long idx = ((long)bh * NCHUNK + c) * 16384 + i;
            prev[idx] = P;
            P = P * dec[c] + states[idx];
        }
    }
}

// ---------------------------------------------------------------- SSD: Y via WMMA
// G = C*B^T -> M (mask+decay, bf16) ; Y = M*X + Cel*P^T + D*x
// grid (NCHUNK, NHEADS, BATCH); dyn LDS: 6*128*SLDT bf16 = 204 KB (CDNA5: 320 KB/WGP)
__global__ __launch_bounds__(256) void k_ssd_y(
    const float* __restrict__ xBC, const float* __restrict__ dt,
    const float* __restrict__ dAcs, const float* __restrict__ prev,
    const float* __restrict__ Dv, float* __restrict__ y)
{
    int c = blockIdx.x, h = blockIdx.y, b = blockIdx.z;
    int kv = h >> 1;
    extern __shared__ unsigned short smu[];
    unsigned short* Cb  = smu;                 // [l][n]
    unsigned short* Cel = smu + 1 * 128 * SLDT;  // [l][n] * exp(cs[l])
    unsigned short* Bb  = smu + 2 * 128 * SLDT;  // [s][n]
    unsigned short* Mb  = smu + 3 * 128 * SLDT;  // [l][s]
    unsigned short* Xt  = smu + 4 * 128 * SLDT;  // [p][s]
    unsigned short* Pb  = smu + 5 * 128 * SLDT;  // [p][n]
    __shared__ float cs[CHUNK], dts[CHUNK];

    const int tid = threadIdx.x;
    long tok0 = (long)b * SEQ + (long)c * CHUNK;
    long bhc  = ((long)(b * NHEADS + h) * NCHUNK + c);

    if (tid < CHUNK) {
        cs[tid]  = dAcs[bhc * CHUNK + tid];
        dts[tid] = dt[(tok0 + tid) * NHEADS + h];
    }
    for (int i = tid; i < 16384; i += 256) {
        int s = i >> 7, n = i & 127;
        Bb[s * SLDT + n] = f2bf(xBC[(tok0 + s) * (long)CONVDIM + DXB + kv * DSTATE + n]);
        Xt[n * SLDT + s] = f2bf(xBC[(tok0 + s) * (long)CONVDIM + kv * DSTATE + n]);
        Pb[s * SLDT + n] = f2bf(prev[bhc * 16384 + i]);   // prev is [p][n]
    }
    __syncthreads();
    for (int i = tid; i < 16384; i += 256) {
        int l = i >> 7, n = i & 127;
        float cv = xBC[(tok0 + l) * (long)CONVDIM + 2 * DXB + h * DSTATE + n];
        Cb[l * SLDT + n]  = f2bf(cv);
        Cel[l * SLDT + n] = f2bf(cv * __expf(cs[l]));
    }
    __syncthreads();

    const int lane = tid & 31, wave = tid >> 5;
    const int m16 = lane & 15, kh = lane >> 4;
    const int lband = wave;
    const int lrow = lband * 16 + m16;

    // ---- phase 1: G tiles -> masked/decayed M (bf16)
    Frag afC[4];
#pragma unroll
    for (int ks = 0; ks < 4; ++ks) afC[ks] = ldsfrag(Cb, lrow, SLDT, ks, kh);

    for (int st = 0; st < 8; ++st) {
        if (st > lband) {
#pragma unroll
            for (int v = 0; v < 8; ++v)
                Mb[(lband * 16 + v + 8 * kh) * SLDT + st * 16 + m16] = 0;
            continue;
        }
        v8f acc = V8F_ZERO;
#pragma unroll
        for (int ks = 0; ks < 4; ++ks) {
            Frag bf = ldsfrag(Bb, st * 16 + m16, SLDT, ks, kh);
            acc = __builtin_amdgcn_wmma_f32_16x16x32_bf16(
                false, afC[ks].v, false, bf.v, (short)0, acc, false, false);
        }
#pragma unroll
        for (int v = 0; v < 8; ++v) {
            int l = lband * 16 + v + 8 * kh;
            int s = st * 16 + m16;
            float val = (s <= l) ? acc[v] * __expf(cs[l] - cs[s]) * dts[s] : 0.0f;
            Mb[l * SLDT + s] = f2bf(val);
        }
    }
    __syncthreads();

    // ---- phase 2: Y = M*X + Cel*P^T + D*x
    Frag afM[4], afCe[4];
#pragma unroll
    for (int ks = 0; ks < 4; ++ks) {
        afM[ks]  = ldsfrag(Mb,  lrow, SLDT, ks, kh);
        afCe[ks] = ldsfrag(Cel, lrow, SLDT, ks, kh);
    }
    float Dh = Dv[h];
    for (int j = 0; j < 8; ++j) {
        v8f acc = V8F_ZERO;
        int pr = j * 16 + m16;
#pragma unroll
        for (int ks = 0; ks < 4; ++ks) {
            Frag bx = ldsfrag(Xt, pr, SLDT, ks, kh);
            acc = __builtin_amdgcn_wmma_f32_16x16x32_bf16(
                false, afM[ks].v, false, bx.v, (short)0, acc, false, false);
        }
#pragma unroll
        for (int ks = 0; ks < 4; ++ks) {
            Frag bp = ldsfrag(Pb, pr, SLDT, ks, kh);
            acc = __builtin_amdgcn_wmma_f32_16x16x32_bf16(
                false, afCe[ks].v, false, bp.v, (short)0, acc, false, false);
        }
#pragma unroll
        for (int v = 0; v < 8; ++v) {
            int l = lband * 16 + v + 8 * kh;
            int p = j * 16 + m16;
            float val = acc[v] + Dh * bf2f(Xt[p * SLDT + l]);
            y[(tok0 + l) * (long)DINNER + h * DSTATE + p] = val;
        }
    }
}

// ---------------------------------------------------------------- elementwise tails
__global__ __launch_bounds__(256) void k_add(
    const float* __restrict__ a, const float* __restrict__ b,
    float* __restrict__ o, long n)
{
    long i = (long)blockIdx.x * 256 + threadIdx.x;
    if (i < n) o[i] = a[i] + b[i];
}

__global__ __launch_bounds__(256) void k_silu_mul_bf16(
    const float* __restrict__ g1, const float* __restrict__ g3,
    unsigned short* __restrict__ o, long n)
{
    long i = (long)blockIdx.x * 256 + threadIdx.x;
    if (i < n) o[i] = f2bf(siluf(g1[i]) * g3[i]);
}

// ---------------------------------------------------------------- host launcher
extern "C" void kernel_launch(void* const* d_in, const int* in_sizes, int n_in,
                              void* d_out, int out_size, void* d_ws, size_t ws_size,
                              hipStream_t stream)
{
    (void)in_sizes; (void)n_in; (void)out_size; (void)ws_size;
    const float* hidden  = (const float*)d_in[0];
    const float* ln1_w   = (const float*)d_in[1];
    const float* W_in    = (const float*)d_in[2];
    const float* conv_w  = (const float*)d_in[3];
    const float* conv_b  = (const float*)d_in[4];
    const float* dt_bias = (const float*)d_in[5];
    const float* A_log   = (const float*)d_in[6];
    const float* Dv      = (const float*)d_in[7];
    const float* norm_w  = (const float*)d_in[8];
    const float* W_out   = (const float*)d_in[9];
    const float* ln2_w   = (const float*)d_in[10];
    const float* w1      = (const float*)d_in[11];
    const float* w3      = (const float*)d_in[12];
    const float* w2      = (const float*)d_in[13];
    float* out = (float*)d_out;

    char* p = (char*)d_ws;
    auto alloc = [&](size_t bytes) -> void* {
        void* r = (void*)p;
        p += (bytes + 255) & ~(size_t)255;
        return r;
    };
    const long M = NTOK;

    unsigned short* hnorm = (unsigned short*)alloc(M * DMODEL * 2);
    unsigned short* Wint  = (unsigned short*)alloc((long)PROJ * DMODEL * 2);
    unsigned short* Woutt = (unsigned short*)alloc((long)DMODEL * DINNER * 2);
    unsigned short* w1t   = (unsigned short*)alloc((long)INTER * DMODEL * 2);
    unsigned short* w3t   = (unsigned short*)alloc((long)INTER * DMODEL * 2);
    unsigned short* w2t   = (unsigned short*)alloc((long)DMODEL * INTER * 2);
    float* zxbcdt = (float*)alloc(M * (long)PROJ * 4);
    float* xBC    = (float*)alloc(M * (long)CONVDIM * 4);
    float* dt     = (float*)alloc(M * NHEADS * 4);
    float* dA     = (float*)alloc(M * NHEADS * 4);
    float* dAcs   = (float*)alloc((long)BATCH * NHEADS * NCHUNK * CHUNK * 4);
    float* states = (float*)alloc((long)BATCH * NHEADS * NCHUNK * 16384 * 4);
    float* prevs  = (float*)alloc((long)BATCH * NHEADS * NCHUNK * 16384 * 4);
    float* ybuf   = (float*)alloc(M * (long)DINNER * 4);
    unsigned short* ygate = (unsigned short*)alloc(M * DINNER * 2);
    float* tmp2   = (float*)alloc(M * (long)DMODEL * 4);
    float* h2     = (float*)alloc(M * (long)DMODEL * 4);
    unsigned short* h2n = (unsigned short*)alloc(M * DMODEL * 2);
    float* g1buf  = (float*)alloc(M * (long)INTER * 4);
    float* g3buf  = (float*)alloc(M * (long)INTER * 4);
    unsigned short* gate = (unsigned short*)alloc(M * INTER * 2);
    float* mlp    = (float*)alloc(M * (long)DMODEL * 4);

    dim3 tb32(32, 8);
    // weight conversions (deterministic, every call)
    k_conv_t_bf16<<<dim3((PROJ + 31) / 32,  (DMODEL + 31) / 32), tb32, 0, stream>>>(W_in,  Wint,  DMODEL, PROJ);
    k_conv_t_bf16<<<dim3((DMODEL + 31) / 32,(DINNER + 31) / 32), tb32, 0, stream>>>(W_out, Woutt, DINNER, DMODEL);
    k_conv_t_bf16<<<dim3((INTER + 31) / 32, (DMODEL + 31) / 32), tb32, 0, stream>>>(w1,    w1t,   DMODEL, INTER);
    k_conv_t_bf16<<<dim3((INTER + 31) / 32, (DMODEL + 31) / 32), tb32, 0, stream>>>(w3,    w3t,   DMODEL, INTER);
    k_conv_t_bf16<<<dim3((DMODEL + 31) / 32,(INTER + 31) / 32),  tb32, 0, stream>>>(w2,    w2t,   INTER,  DMODEL);

    // 1) rmsnorm -> bf16
    k_rmsnorm_bf16<<<NTOK, 256, 0, stream>>>(hidden, ln1_w, hnorm, DMODEL);
    // 2) in-projection GEMM
    k_gemm_bf16<<<dim3((PROJ + BN - 1) / BN, M / BM), 256, 0, stream>>>(hnorm, Wint, zxbcdt, M, PROJ, DMODEL);
    // 3) conv + silu
    k_conv_silu<<<(unsigned)((M * (long)CONVDIM) / 256), 256, 0, stream>>>(zxbcdt, conv_w, conv_b, xBC);
    // 4) dt / dA
    k_dtprep<<<(NTOK * NHEADS) / 256, 256, 0, stream>>>(zxbcdt, dt_bias, A_log, dt, dA);
    // 5) SSD (WMMA chunk matmuls)
    k_ssd_states<<<dim3(NCHUNK, NHEADS, BATCH), 256, 2 * 128 * SLDT * 2, stream>>>(xBC, dt, dA, states, dAcs);
    k_ssd_scan<<<BATCH * NHEADS, 256, 0, stream>>>(states, dAcs, prevs);
    k_ssd_y<<<dim3(NCHUNK, NHEADS, BATCH), 256, 6 * 128 * SLDT * 2, stream>>>(xBC, dt, dAcs, prevs, Dv, ybuf);
    // 6) gated norm -> bf16
    k_gatenorm<<<NTOK, 256, 0, stream>>>(ybuf, zxbcdt, norm_w, ygate);
    // 7) out-projection + residual
    k_gemm_bf16<<<dim3(DMODEL / BN, M / BM), 256, 0, stream>>>(ygate, Woutt, tmp2, M, DMODEL, DINNER);
    k_add<<<(unsigned)((M * DMODEL) / 256), 256, 0, stream>>>(hidden, tmp2, h2, M * (long)DMODEL);
    // 8) MLP
    k_rmsnorm_bf16<<<NTOK, 256, 0, stream>>>(h2, ln2_w, h2n, DMODEL);
    k_gemm_bf16<<<dim3(INTER / BN, M / BM), 256, 0, stream>>>(h2n, w1t, g1buf, M, INTER, DMODEL);
    k_gemm_bf16<<<dim3(INTER / BN, M / BM), 256, 0, stream>>>(h2n, w3t, g3buf, M, INTER, DMODEL);
    k_silu_mul_bf16<<<(unsigned)((M * (long)INTER) / 256), 256, 0, stream>>>(g1buf, g3buf, gate, M * (long)INTER);
    k_gemm_bf16<<<dim3(DMODEL / BN, M / BM), 256, 0, stream>>>(gate, w2t, mlp, M, DMODEL, INTER);
    // 9) final residual
    k_add<<<(unsigned)((M * DMODEL) / 256), 256, 0, stream>>>(h2, mlp, out, M * (long)DMODEL);
}
